// ResourceEmbeddingLayer_74217034875539
// MI455X (gfx1250) — matrix operand order repaired and checked by reading, compiled, not run on hardware
//
#include <hip/hip_runtime.h>

typedef __bf16 bf16_t;
typedef __attribute__((ext_vector_type(16))) __bf16 v16bf;
typedef __attribute__((ext_vector_type(8)))  __bf16 v8bf;
typedef __attribute__((ext_vector_type(8)))  float  v8f;

#define EMB   256
#define KSELF 128
#define KOPS  192
#define KEDGE 64
#define MT    4            // edge tiles (of 16) per block in k_egemm

static __device__ __forceinline__ float leaky02(float x) { return x > 0.f ? x : 0.2f * x; }
static __device__ __forceinline__ float eluf(float x)    { return x > 0.f ? x : (__expf(x) - 1.f); }

static __device__ __forceinline__ unsigned short cvt_bf16(float f) {
  unsigned u = __float_as_uint(f);
  u += 0x7FFFu + ((u >> 16) & 1u);          // round-to-nearest-even
  return (unsigned short)(u >> 16);
}
static __device__ __forceinline__ unsigned enc_ord(float f) {
  unsigned u = __float_as_uint(f);
  return (u & 0x80000000u) ? ~u : (u | 0x80000000u);
}
static __device__ __forceinline__ float dec_ord(unsigned u) {
  return (u & 0x80000000u) ? __uint_as_float(u ^ 0x80000000u) : __uint_as_float(~u);
}

static __device__ __forceinline__ v8f wmma_bf16(v16bf a, v16bf b, v8f c) {
  return __builtin_amdgcn_wmma_f32_16x16x32_bf16(false, a, false, b, (short)0, c, false, false);
}
// A-matrix 16x32 bf16 fragment: lane holds row M=lane&15,
// 16 contiguous K values starting at kb + (lane>>4)*16.
static __device__ __forceinline__ v16bf load_a_frag(const bf16_t* p) {
  return *(const v16bf*)(p);
}
// B-matrix 32x16 fragment: lane holds column N=lane&15, 16 contiguous K
// values starting at kb + (lane>>4)*16 (column-of-Wt == row of W_op).
static __device__ __forceinline__ v16bf load_b_frag(const bf16_t* p) {
  return *(const v16bf*)(p);
}

// ---- prep: fp32 -> bf16 weight copies ----
__global__ void k_convert(const float* Wself, const float* Wop,
                          unsigned short* WselfB, unsigned short* WopB) {
  int i = blockIdx.x * blockDim.x + threadIdx.x;
  const int nself = EMB * KSELF, nop = EMB * EMB;
  if (i < nself)            WselfB[i] = cvt_bf16(Wself[i]);
  else if (i < nself + nop) WopB[i - nself] = cvt_bf16(Wop[i - nself]);
}

// ---- w2 = W_op^T @ a_op[EMB:], avec = a_self[:EMB]+a_self[EMB:], init reductions ----
__global__ void k_wvec(const float* Wop, const float* a_self, const float* a_op,
                       float* w2, float* avec, unsigned* maxb, float* Zp) {
  int k = threadIdx.x;  // EMB threads
  float s = 0.f;
  for (int n = 0; n < EMB; ++n) s += Wop[n * EMB + k] * a_op[EMB + n];
  w2[k]   = s;
  avec[k] = a_self[k] + a_self[EMB + k];
  if (k == 0) { *maxb = 0u; *Zp = 0.f; }
}

// ---- h = resources @ W_self^T via bf16 WMMA ----
__global__ void __launch_bounds__(256) k_hgemm(const float* resources,
                                               const unsigned short* WselfB,
                                               float* h, int R) {
  __shared__ __align__(32) unsigned short Xl[16 * KSELF];
  const int tid = threadIdx.x, tile = blockIdx.x;
  for (int i = tid; i < 16 * KSELF; i += 256) {
    int m = i >> 7, k = i & (KSELF - 1);
    int r = tile * 16 + m;
    Xl[i] = cvt_bf16(r < R ? resources[(size_t)r * KSELF + k] : 0.f);
  }
  __syncthreads();
  const int lane = tid & 31, wave = tid >> 5;
  const int n0 = wave * 32, m16 = lane & 15, hi = lane >> 4;
  const bf16_t* X = (const bf16_t*)Xl;
  const bf16_t* W = (const bf16_t*)WselfB;
  v8f acc0 = {}; v8f acc1 = {};
#pragma unroll
  for (int kb = 0; kb < KSELF; kb += 32) {
    v16bf a  = load_a_frag(X + m16 * KSELF + kb + hi * 16);
    v16bf b0 = load_b_frag(W + (size_t)(n0 + m16) * KSELF + kb + hi * 16);
    v16bf b1 = load_b_frag(W + (size_t)(n0 + 16 + m16) * KSELF + kb + hi * 16);
    acc0 = wmma_bf16(a, b0, acc0);
    acc1 = wmma_bf16(a, b1, acc1);
  }
  float* hrow = h + (size_t)(tile * 16 + hi * 8) * EMB + n0 + m16;
  if (tile * 16 + 16 <= R) {            // full tile: branchless stores
#pragma unroll
    for (int v = 0; v < 8; ++v) {
      hrow[(size_t)v * EMB]      = acc0[v];
      hrow[(size_t)v * EMB + 16] = acc1[v];
    }
  } else {
#pragma unroll
    for (int v = 0; v < 8; ++v) {
      int r = tile * 16 + hi * 8 + v;
      if (r < R) {
        h[(size_t)r * EMB + n0 + m16]      = acc0[v];
        h[(size_t)r * EMB + n0 + 16 + m16] = acc1[v];
      }
    }
  }
}

// ---- per-resource dots: hdot = h.a_op_lo ; sscore = leaky(h.avec) ----
__global__ void k_rdots(const float* h, const float* a_op, const float* avec,
                        float* hdot, float* sscore, int R) {
  int gw = (blockIdx.x * blockDim.x + threadIdx.x) >> 5;
  int lane = threadIdx.x & 31;
  if (gw >= R) return;
  const float* hr = h + (size_t)gw * EMB;
  float s1 = 0.f, s2 = 0.f;
  for (int i = lane; i < EMB; i += 32) {
    float hv = hr[i];
    s1 += hv * a_op[i];
    s2 += hv * avec[i];
  }
  for (int m = 16; m >= 1; m >>= 1) {
    s1 += __shfl_xor(s1, m, 32);
    s2 += __shfl_xor(s2, m, 32);
  }
  if (lane == 0) { hdot[gw] = s1; sscore[gw] = leaky02(s2); }
}

// ---- per-operation dot: opdot = operations . w2[:192] ----
__global__ void k_opdot(const float* operations, const float* w2, float* opdot, int O) {
  int gw = (blockIdx.x * blockDim.x + threadIdx.x) >> 5;
  int lane = threadIdx.x & 31;
  if (gw >= O) return;
  const float* op = operations + (size_t)gw * KOPS;
  float s = 0.f;
#pragma unroll
  for (int j = 0; j < KOPS / 32; ++j) s += op[lane + 32 * j] * w2[lane + 32 * j];
  for (int m = 16; m >= 1; m >>= 1) s += __shfl_xor(s, m, 32);
  if (lane == 0) opdot[gw] = s;
}

// ---- per-edge score: cscore = leaky(hdot[dst] + opdot[src] + edge_attr.w2[192:]) ----
__global__ void k_escore(const float* edge_attr, const int* esrc, const int* edst,
                         const float* hdot, const float* opdot, const float* w2,
                         float* cscore, int E) {
  int gw = (blockIdx.x * blockDim.x + threadIdx.x) >> 5;
  int lane = threadIdx.x & 31;
  if (gw >= E) return;
  const float* ea  = edge_attr + (size_t)gw * KEDGE;
  const float* w2e = w2 + KOPS;
  float s = ea[lane] * w2e[lane] + ea[lane + 32] * w2e[lane + 32];
  for (int m = 16; m >= 1; m >>= 1) s += __shfl_xor(s, m, 32);
  if (lane == 0)
    cscore[gw] = leaky02(hdot[edst[gw]] + opdot[esrc[gw]] + s);
}

// ---- global softmax reductions ----
__global__ void k_max(const float* sscore, const float* cscore, int R, int E, unsigned* maxb) {
  int tid = blockIdx.x * blockDim.x + threadIdx.x;
  int stride = gridDim.x * blockDim.x, N = R + E;
  float mx = -3.4e38f;
  for (int i = tid; i < N; i += stride)
    mx = fmaxf(mx, (i < R) ? sscore[i] : cscore[i - R]);
  for (int m = 16; m >= 1; m >>= 1) mx = fmaxf(mx, __shfl_xor(mx, m, 32));
  if ((threadIdx.x & 31) == 0) atomicMax(maxb, enc_ord(mx));
}
__global__ void k_sum(const float* sscore, const float* cscore, int R, int E,
                      const unsigned* maxb, float* Zp) {
  float mx = dec_ord(*maxb);
  int tid = blockIdx.x * blockDim.x + threadIdx.x;
  int stride = gridDim.x * blockDim.x, N = R + E;
  float s = 0.f;
  for (int i = tid; i < N; i += stride)
    s += __expf(((i < R) ? sscore[i] : cscore[i - R]) - mx);
  for (int m = 16; m >= 1; m >>= 1) s += __shfl_xor(s, m, 32);
  if ((threadIdx.x & 31) == 0) atomicAdd(Zp, s);
}

__global__ void k_zero(float* p, int n) {
  int i = blockIdx.x * blockDim.x + threadIdx.x;
  if (i < n) p[i] = 0.f;
}

// ---- big fused kernel: ops_e (WMMA, 4 M-tiles/block for B reuse) * weight -> atomic segment-sum ----
__global__ void __launch_bounds__(256) k_egemm(const float* operations, const float* edge_attr,
                                               const int* esrc, const int* edst,
                                               const float* cscore, const unsigned short* WopB,
                                               const unsigned* maxb, const float* Zp,
                                               float* accum, int E) {
  __shared__ __align__(32) unsigned short Xl[MT * 16 * EMB];   // 32 KB
  __shared__ float wl[MT * 16];
  __shared__ int   sl[MT * 16], dl[MT * 16];
  const int tid = threadIdx.x;
  const size_t ebase = (size_t)blockIdx.x * (MT * 16);
  if (tid < MT * 16) {
    size_t e = ebase + tid;
    if (e < (size_t)E) {
      sl[tid] = esrc[e];
      dl[tid] = edst[e];
      wl[tid] = __expf(cscore[e] - dec_ord(*maxb)) / *Zp;
    } else { sl[tid] = 0; dl[tid] = 0; wl[tid] = 0.f; }
  }
  __syncthreads();
  for (int i = tid; i < MT * 16 * EMB; i += 256) {   // gather + convert X tiles
    int m = i >> 8, k = i & (EMB - 1);
    size_t e = ebase + m;
    float v = 0.f;
    if (e < (size_t)E) v = (k < KOPS) ? operations[(size_t)sl[m] * KOPS + k]
                                      : edge_attr[e * KEDGE + (k - KOPS)];
    Xl[i] = cvt_bf16(v);
  }
  __syncthreads();
  const int lane = tid & 31, wave = tid >> 5;
  const int n0 = wave * 32, m16 = lane & 15, hi = lane >> 4;
  const bf16_t* X = (const bf16_t*)Xl;
  const bf16_t* W = (const bf16_t*)WopB;
  v8f acc[MT][2] = {};
#pragma unroll
  for (int kb = 0; kb < EMB; kb += 32) {
    v16bf b0 = load_b_frag(W + (size_t)(n0 + m16) * EMB + kb + hi * 16);
    v16bf b1 = load_b_frag(W + (size_t)(n0 + 16 + m16) * EMB + kb + hi * 16);
#pragma unroll
    for (int t = 0; t < MT; ++t) {
      v16bf a = load_a_frag(X + (t * 16 + m16) * EMB + kb + hi * 16);
      acc[t][0] = wmma_bf16(a, b0, acc[t][0]);
      acc[t][1] = wmma_bf16(a, b1, acc[t][1]);
    }
  }
#pragma unroll
  for (int t = 0; t < MT; ++t) {
#pragma unroll
    for (int v = 0; v < 8; ++v) {
      int m = t * 16 + hi * 8 + v;
      float w = wl[m];
      int   d = dl[m];
      unsafeAtomicAdd(&accum[(size_t)d * EMB + n0 + m16],      acc[t][0][v] * w);
      unsafeAtomicAdd(&accum[(size_t)d * EMB + n0 + 16 + m16], acc[t][1][v] * w);
    }
  }
}

// ---- out = elu(norm_self * h + accum) ----
__global__ void k_final(const float* h, const float* accum, const float* sscore,
                        const unsigned* maxb, const float* Zp, float* out, int R) {
  int i = blockIdx.x * blockDim.x + threadIdx.x;
  if (i >= R * EMB) return;
  int r = i >> 8;
  float ns = __expf(sscore[r] - dec_ord(*maxb)) / *Zp;
  out[i] = eluf(ns * h[i] + accum[i]);
}

extern "C" void kernel_launch(void* const* d_in, const int* in_sizes, int n_in,
                              void* d_out, int out_size, void* d_ws, size_t ws_size,
                              hipStream_t stream) {
  const float* resources  = (const float*)d_in[0];
  const float* operations = (const float*)d_in[1];
  const float* edge_attr  = (const float*)d_in[2];
  const float* W_self     = (const float*)d_in[3];
  const float* W_op       = (const float*)d_in[4];
  const float* a_self     = (const float*)d_in[5];
  const float* a_op       = (const float*)d_in[6];
  const int*   esrc       = (const int*)d_in[7];
  const int*   edst       = (const int*)d_in[8];
  float* out = (float*)d_out;

  const int R = in_sizes[0] / KSELF;
  const int O = in_sizes[1] / KOPS;
  const int E = in_sizes[2] / KEDGE;

  float* ws = (float*)d_ws;
  size_t off = 0;
  auto align16 = [](size_t x) { return (x + 15) & ~(size_t)15; };  // 64B in floats
  float* h      = ws + off; off += (size_t)R * EMB;
  float* hdot   = ws + off; off += R;
  float* sscore = ws + off; off += R;
  float* opdot  = ws + off; off += O;
  float* cscore = ws + off; off += E;
  float* w2     = ws + off; off += EMB;
  float* avec   = ws + off; off += EMB;
  unsigned* maxb = (unsigned*)(ws + off); off += 1;
  float* Zp     = ws + off; off += 1;
  float* accum  = ws + off; off += (size_t)R * EMB;
  off = align16(off);                                   // 32B-align bf16 weights
  unsigned short* WselfB = (unsigned short*)(ws + off); off += (EMB * KSELF) / 2;
  unsigned short* WopB   = (unsigned short*)(ws + off); off += (EMB * EMB) / 2;

  const int ncv = EMB * KSELF + EMB * EMB;
  k_convert<<<(ncv + 255) / 256, 256, 0, stream>>>(W_self, W_op, WselfB, WopB);
  k_wvec<<<1, EMB, 0, stream>>>(W_op, a_self, a_op, w2, avec, maxb, Zp);
  k_hgemm<<<(R + 15) / 16, 256, 0, stream>>>(resources, WselfB, h, R);
  k_rdots<<<(R + 7) / 8, 256, 0, stream>>>(h, a_op, avec, hdot, sscore, R);
  k_opdot<<<(O + 7) / 8, 256, 0, stream>>>(operations, w2, opdot, O);
  k_escore<<<(E + 7) / 8, 256, 0, stream>>>(edge_attr, esrc, edst, hdot, opdot, w2, cscore, E);
  k_max<<<1024, 256, 0, stream>>>(sscore, cscore, R, E, maxb);
  k_sum<<<1024, 256, 0, stream>>>(sscore, cscore, R, E, maxb, Zp);
  k_zero<<<(R * EMB + 255) / 256, 256, 0, stream>>>(accum, R * EMB);
  k_egemm<<<(E + MT * 16 - 1) / (MT * 16), 256, 0, stream>>>(operations, edge_attr, esrc, edst,
                                                             cscore, WopB, maxb, Zp, accum, E);
  k_final<<<(R * EMB + 255) / 256, 256, 0, stream>>>(h, accum, sscore, maxb, Zp, out, R);
}
